// _FusionAttention_43258910605762
// MI455X (gfx1250) — matrix operand order repaired and checked by reading, not compiled, not run on hardware
//
#include <hip/hip_runtime.h>
#include <math.h>

#define D_MODEL 256
#define NSEQ    4096
#define NBATCH  32
#define LDS_PITCH 264   // 256 + 8 halves: shifts each row by 4 banks -> conflict-free b128 reads

typedef __attribute__((ext_vector_type(16))) _Float16 v16h;
typedef __attribute__((ext_vector_type(8)))  _Float16 v8h;
typedef __attribute__((ext_vector_type(8)))  float    v8f;

__device__ __forceinline__ v16h combine16(const _Float16* p0, const _Float16* p1) {
  const v8h lo = *(const v8h*)p0;
  const v8h hi = *(const v8h*)p1;
  return __builtin_shufflevector(lo, hi, 0,1,2,3,4,5,6,7,8,9,10,11,12,13,14,15);
}

// Pre-pass: convert the three 256x256 fp32 weight matrices to f16 once.
__global__ __launch_bounds__(256) void cvt_weights_kernel(
    const float* __restrict__ qw, const float* __restrict__ kw,
    const float* __restrict__ vw,
    _Float16* __restrict__ wq, _Float16* __restrict__ wk, _Float16* __restrict__ wv)
{
  const int i = blockIdx.x * 256 + threadIdx.x;  // 0 .. 65535
  wq[i] = (_Float16)qw[i];
  wk[i] = (_Float16)kw[i];
  wv[i] = (_Float16)vw[i];
}

// One block = one 32-row tile of n for one batch (two 16-row WMMA row-tiles).
// 8 waves; wave w owns e-column tiles {2w, 2w+1}. Each B fragment feeds both
// row-tiles (2 WMMAs), halving weight fragment traffic vs 16-row blocks.
__global__ __launch_bounds__(256) void qkv_score_kernel(
    const float* __restrict__ temporal, const float* __restrict__ spatial,
    const _Float16* __restrict__ wq, const float* __restrict__ qb,
    const _Float16* __restrict__ wk, const float* __restrict__ kb,
    const _Float16* __restrict__ wv, const float* __restrict__ vb,
    float* __restrict__ v_ws, float* __restrict__ score_ws)
{
  const int tilesPerB = NSEQ / 32;
  const int b    = blockIdx.x / tilesPerB;
  const int n0   = (blockIdx.x % tilesPerB) * 32;
  const int wave = threadIdx.x >> 5;
  const int lane = threadIdx.x & 31;
  const int h    = lane >> 4;   // half-wave select
  const int ln   = lane & 15;   // row (A) / column (B,C,D) index

  // Stage the two 32x256 A tiles into LDS as f16 (converted once per block).
  __shared__ _Float16 Tt[32][LDS_PITCH];
  __shared__ _Float16 Ts[32][LDS_PITCH];
  {
    const int row = threadIdx.x >> 3;        // 0..31
    const int c0  = (threadIdx.x & 7) * 32;  // 0,32,...,224
    const float* tg = temporal + (size_t)(b * NSEQ + n0 + row) * D_MODEL + c0;
    const float* sg = spatial  + (size_t)(b * NSEQ + n0 + row) * D_MODEL + c0;
    #pragma unroll
    for (int j = 0; j < 8; ++j) {
      const float4 tv = ((const float4*)tg)[j];
      const float4 sv = ((const float4*)sg)[j];
      Tt[row][c0 + 4 * j + 0] = (_Float16)tv.x;
      Tt[row][c0 + 4 * j + 1] = (_Float16)tv.y;
      Tt[row][c0 + 4 * j + 2] = (_Float16)tv.z;
      Tt[row][c0 + 4 * j + 3] = (_Float16)tv.w;
      Ts[row][c0 + 4 * j + 0] = (_Float16)sv.x;
      Ts[row][c0 + 4 * j + 1] = (_Float16)sv.y;
      Ts[row][c0 + 4 * j + 2] = (_Float16)sv.z;
      Ts[row][c0 + 4 * j + 3] = (_Float16)sv.w;
    }
  }
  __syncthreads();

  const int e[2] = { (wave * 2 + 0) * 16 + ln, (wave * 2 + 1) * 16 + ln };
  const _Float16* qwr[2] = { wq + (size_t)e[0] * D_MODEL, wq + (size_t)e[1] * D_MODEL };
  const _Float16* kwr[2] = { wk + (size_t)e[0] * D_MODEL, wk + (size_t)e[1] * D_MODEL };
  const _Float16* vwr[2] = { wv + (size_t)e[0] * D_MODEL, wv + (size_t)e[1] * D_MODEL };

  const _Float16* trow[2] = { &Tt[ln][0], &Tt[16 + ln][0] };
  const _Float16* srow[2] = { &Ts[ln][0], &Ts[16 + ln][0] };

  v8f accq[2][2], acck[2][2], accv[2][2];
  #pragma unroll
  for (int ci = 0; ci < 2; ++ci)
    #pragma unroll
    for (int rt = 0; rt < 2; ++rt) { accq[ci][rt] = (v8f){}; acck[ci][rt] = (v8f){}; accv[ci][rt] = (v8f){}; }

  #pragma unroll
  for (int ks = 0; ks < 8; ++ks) {
    const int kb0 = ks * 32;
    // A fragments for both row-tiles: K chunks [kb0+16h,+8) and [kb0+16h+8,+8).
    v16h at[2], as[2];
    #pragma unroll
    for (int rt = 0; rt < 2; ++rt) {
      at[rt] = combine16(trow[rt] + kb0 + 16 * h, trow[rt] + kb0 + 16 * h + 8);
      as[rt] = combine16(srow[rt] + kb0 + 16 * h, srow[rt] + kb0 + 16 * h + 8);
    }
    // B fragments: 16 contiguous f16 K values [kb0+16h,+16) of column e.
    #pragma unroll
    for (int ci = 0; ci < 2; ++ci) {
      const v16h bq = *(const v16h*)(qwr[ci] + kb0 + 16 * h);
      const v16h bk = *(const v16h*)(kwr[ci] + kb0 + 16 * h);
      const v16h bv = *(const v16h*)(vwr[ci] + kb0 + 16 * h);
      #pragma unroll
      for (int rt = 0; rt < 2; ++rt) {
        accq[ci][rt] = __builtin_amdgcn_wmma_f32_16x16x32_f16(false, at[rt], false, bq, (short)0, accq[ci][rt], false, false);
        acck[ci][rt] = __builtin_amdgcn_wmma_f32_16x16x32_f16(false, as[rt], false, bk, (short)0, acck[ci][rt], false, false);
        accv[ci][rt] = __builtin_amdgcn_wmma_f32_16x16x32_f16(false, as[rt], false, bv, (short)0, accv[ci][rt], false, false);
      }
    }
  }

  // C/D layout: VGPR r, lane -> element (m = rt*16 + r + 8h, e = ctile*16 + ln).
  float sloc[2][8];
  #pragma unroll
  for (int rt = 0; rt < 2; ++rt)
    #pragma unroll
    for (int r = 0; r < 8; ++r) sloc[rt][r] = 0.0f;

  #pragma unroll
  for (int ci = 0; ci < 2; ++ci) {
    const float qbe = qb[e[ci]], kbe = kb[e[ci]], vbe = vb[e[ci]];
    #pragma unroll
    for (int rt = 0; rt < 2; ++rt) {
      #pragma unroll
      for (int r = 0; r < 8; ++r) {
        sloc[rt][r] += (accq[ci][rt][r] + qbe) * (acck[ci][rt][r] + kbe);
        v_ws[(size_t)(b * NSEQ + n0 + rt * 16 + r + 8 * h) * D_MODEL + e[ci]] = accv[ci][rt][r] + vbe;
      }
    }
  }

  // Reduce score partials across the 16 lanes of each half-wave.
  #pragma unroll
  for (int rt = 0; rt < 2; ++rt) {
    #pragma unroll
    for (int r = 0; r < 8; ++r) {
      float s = sloc[rt][r];
      s += __shfl_xor(s, 1, 32);
      s += __shfl_xor(s, 2, 32);
      s += __shfl_xor(s, 4, 32);
      s += __shfl_xor(s, 8, 32);
      sloc[rt][r] = s;
    }
  }

  __shared__ float sred[8][2][16];
  if (ln == 0) {
    #pragma unroll
    for (int rt = 0; rt < 2; ++rt)
      #pragma unroll
      for (int r = 0; r < 8; ++r) sred[wave][rt][h * 8 + r] = sloc[rt][r];
  }
  __syncthreads();
  if (threadIdx.x < 32) {
    const int rt  = threadIdx.x >> 4;
    const int row = threadIdx.x & 15;
    float s = 0.0f;
    #pragma unroll
    for (int w = 0; w < 8; ++w) s += sred[w][rt][row];
    score_ws[b * NSEQ + n0 + rt * 16 + row] = s;
  }
}

// One block per batch: softmax over the N=4096 scores, rewritten in place.
__global__ __launch_bounds__(256) void softmax_kernel(float* __restrict__ score)
{
  float* s = score + blockIdx.x * NSEQ;
  const int t = threadIdx.x;
  float vals[16];
  float mx = -INFINITY;
  #pragma unroll
  for (int i = 0; i < 16; ++i) { vals[i] = s[t + i * 256]; mx = fmaxf(mx, vals[i]); }

  __shared__ float red[256];
  red[t] = mx; __syncthreads();
  for (int off = 128; off > 0; off >>= 1) {
    if (t < off) red[t] = fmaxf(red[t], red[t + off]);
    __syncthreads();
  }
  mx = red[0]; __syncthreads();

  float sum = 0.0f;
  #pragma unroll
  for (int i = 0; i < 16; ++i) { vals[i] = __expf(vals[i] - mx); sum += vals[i]; }
  red[t] = sum; __syncthreads();
  for (int off = 128; off > 0; off >>= 1) {
    if (t < off) red[t] += red[t + off];
    __syncthreads();
  }
  const float inv = 1.0f / red[0];
  #pragma unroll
  for (int i = 0; i < 16; ++i) s[t + i * 256] = vals[i] * inv;
}

// out = temporal + attn[b,n] * v
__global__ __launch_bounds__(256) void combine_kernel(
    const float* __restrict__ temporal, const float* __restrict__ v_ws,
    const float* __restrict__ attn, float* __restrict__ out)
{
  const size_t total4 = (size_t)NBATCH * NSEQ * D_MODEL / 4;
  for (size_t i = (size_t)blockIdx.x * blockDim.x + threadIdx.x; i < total4;
       i += (size_t)gridDim.x * blockDim.x) {
    const float a  = attn[i >> 6];             // 64 float4 per (b, n)
    const float4 tv = ((const float4*)temporal)[i];
    const float4 vv = ((const float4*)v_ws)[i];
    float4 o;
    o.x = tv.x + a * vv.x;
    o.y = tv.y + a * vv.y;
    o.z = tv.z + a * vv.z;
    o.w = tv.w + a * vv.w;
    ((float4*)out)[i] = o;
  }
}

extern "C" void kernel_launch(void* const* d_in, const int* in_sizes, int n_in,
                              void* d_out, int out_size, void* d_ws, size_t ws_size,
                              hipStream_t stream) {
  const float* temporal = (const float*)d_in[0];
  const float* spatial  = (const float*)d_in[1];
  const float* qw = (const float*)d_in[2];
  const float* qb = (const float*)d_in[3];
  const float* kw = (const float*)d_in[4];
  const float* kb = (const float*)d_in[5];
  const float* vw = (const float*)d_in[6];
  const float* vb = (const float*)d_in[7];

  float*    v_ws     = (float*)d_ws;
  float*    score_ws = v_ws + (size_t)NBATCH * NSEQ * D_MODEL;
  _Float16* wq       = (_Float16*)(score_ws + (size_t)NBATCH * NSEQ);
  _Float16* wk       = wq + (size_t)D_MODEL * D_MODEL;
  _Float16* wv       = wk + (size_t)D_MODEL * D_MODEL;

  cvt_weights_kernel<<<(D_MODEL * D_MODEL) / 256, 256, 0, stream>>>(qw, kw, vw, wq, wk, wv);
  qkv_score_kernel<<<NBATCH * (NSEQ / 32), 256, 0, stream>>>(
      temporal, spatial, wq, qb, wk, kb, wv, vb, v_ws, score_ws);
  softmax_kernel<<<NBATCH, 256, 0, stream>>>(score_ws);
  combine_kernel<<<4096, 256, 0, stream>>>(temporal, v_ws, score_ws, (float*)d_out);
}
